// BertCrossAttention_6073083757131
// MI455X (gfx1250) — matrix operand order repaired and checked by reading, not compiled, not run on hardware
//
#include <hip/hip_runtime.h>

// ---------------------------------------------------------------------------
// BERT cross-attention for MI455X (gfx1250, wave32, WMMA).
// - All matmuls: v_wmma_f32_16x16x32_f16 (f32 accumulate).
// - GEMM A-panels staged via Tensor Data Mover, double-buffered
//   (tensor_load_to_lds overlapped with WMMA via s_wait_tensorcnt 1).
// - probs (201 MB f32, write-once) stored non-temporally so the hot ~86 MB
//   working set stays resident in the 192 MB L2.
// ---------------------------------------------------------------------------

typedef __attribute__((ext_vector_type(16))) _Float16 v16h;
typedef __attribute__((ext_vector_type(8)))  _Float16 v8h;
typedef __attribute__((ext_vector_type(4)))  _Float16 v4h;
typedef __attribute__((ext_vector_type(8)))  float    v8f;
typedef __attribute__((ext_vector_type(4)))  float    f32x4;
typedef unsigned int u32;
typedef __attribute__((ext_vector_type(4))) u32 u32x4;
typedef __attribute__((ext_vector_type(8))) int i32x8;
typedef __attribute__((ext_vector_type(4))) int i32x4;

#define HID   768
#define HEADS 12
#define HDIM  64
#define LQ    512
#define LK    1024
#define BATCH 8

// ---- WMMA helper: D = A(16x32 f16) * B(32x16 f16) + C(16x16 f32) ----------
__device__ __forceinline__ v8f wmma16(v16h a, v16h b, v8f c) {
  return __builtin_amdgcn_wmma_f32_16x16x32_f16(
      /*neg_a=*/false, a, /*neg_b=*/false, b,
      /*c_mod=*/(short)0, c, /*reuse_a=*/false, /*reuse_b=*/false);
}

// A fragment (16x32, row m = lane%16): per ISA 7.12.2, lane-half hlf=lane>>4
// holds K = {hlf*8..hlf*8+7, 16+hlf*8..16+hlf*8+7} -> two contiguous 16B loads.
__device__ __forceinline__ v16h load_a_frag(const _Float16* base, int hlf) {
  v8h lo = *(const v8h*)(base + hlf * 8);
  v8h hi = *(const v8h*)(base + 16 + hlf * 8);
  return __builtin_shufflevector(lo, hi, 0,1,2,3,4,5,6,7,8,9,10,11,12,13,14,15);
}
// B fragment (32x16, col n = lane%16): lane holds K = {hlf*8..hlf*8+7, 16+hlf*8..16+hlf*8+7}
// of "column n" == row n of the [N][K] weight matrix -> two 16B loads.
__device__ __forceinline__ v16h load_b_frag(const _Float16* rowbase, int hlf) {
  v8h lo = *(const v8h*)(rowbase + hlf * 8);
  v8h hi = *(const v8h*)(rowbase + 16 + hlf * 8);
  return __builtin_shufflevector(lo, hi, 0,1,2,3,4,5,6,7,8,9,10,11,12,13,14,15);
}

// ---- Tensor Data Mover: 2-D tile (rows x k) global -> LDS, f16 elements ---
__device__ __forceinline__ void tdm_load_2d(u32 lds_off, const void* gaddr,
                                            u32 tile_k, u32 tile_rows,
                                            u32 tensor_k, u32 tensor_rows,
                                            u32 row_stride_elems) {
  unsigned long long ga = (unsigned long long)gaddr;
  u32x4 g0;
  g0.x = 1u;                                   // count=1 (valid user D#)
  g0.y = lds_off;                              // lds_addr (bytes)
  g0.z = (u32)(ga & 0xFFFFFFFFull);            // global_addr[31:0]
  g0.w = (u32)((ga >> 32) & 0x1FFFFFFull) | (2u << 30);  // addr[56:32], type=2
  i32x8 g1;
  g1[0] = (int)(1u << 16);                                     // data_size=2B
  g1[1] = (int)((tensor_k & 0xFFFFu) << 16);                   // dim0[15:0]
  g1[2] = (int)(((tensor_k >> 16) & 0xFFFFu) |
                ((tensor_rows & 0xFFFFu) << 16));              // dim0 hi, dim1 lo
  g1[3] = (int)(((tensor_rows >> 16) & 0xFFFFu) |
                ((tile_k & 0xFFFFu) << 16));                   // dim1 hi, tile0
  g1[4] = (int)(tile_rows & 0xFFFFu);                          // tile1, tile2=0
  g1[5] = (int)row_stride_elems;                               // dim0_stride lo
  g1[6] = 0;
  g1[7] = 0;
  i32x4 z4 = {0, 0, 0, 0};
#if __has_include(<hip/amd_detail/amd_gfx1250_TDM.h>)
  i32x8 z8 = {0, 0, 0, 0, 0, 0, 0, 0};
  __builtin_amdgcn_tensor_load_to_lds(g0, g1, z4, z4, z8, 0);
#else
  __builtin_amdgcn_tensor_load_to_lds(g0, g1, z4, z4, 0);
#endif
}

// ---------------------------------------------------------------------------
// Kernel 1: f32 -> f16 convert, vectorized 4-wide (b128 in, b64 out)
// ---------------------------------------------------------------------------
__global__ void k_f32_to_f16(const float* __restrict__ src,
                             _Float16* __restrict__ dst, int n4) {
  int i = blockIdx.x * blockDim.x + threadIdx.x;
  const int stride = gridDim.x * blockDim.x;
  const f32x4* s4 = (const f32x4*)src;
  v4h* d4 = (v4h*)dst;
  for (; i < n4; i += stride) {
    f32x4 v = s4[i];
    v4h h = {(_Float16)v.x, (_Float16)v.y, (_Float16)v.z, (_Float16)v.w};
    d4[i] = h;
  }
}

// ---------------------------------------------------------------------------
// Kernel 2: WMMA GEMM  Y[M][768] = X[M][768] @ W[768][768]^T + bias
// Block = 128 threads (4 waves), 64x64 output tile. A-panel chunks
// (64 rows x 64 k) staged by the TDM with DOUBLE BUFFERING: chunk i+1's
// tensor_load_to_lds is issued before computing chunk i; s_wait_tensorcnt(1)
// (TDM completes in order per wave) releases chunk i while i+1 streams.
// MODE 0: f16 out to [B,H,L,64]. MODE 1: f16 out transposed to [B,H,64,Lk].
// MODE 2: f32 out + bias + residual (row major).
// ---------------------------------------------------------------------------
template <int MODE>
__global__ __launch_bounds__(128) void k_gemm_wmma(
    const _Float16* __restrict__ A, const _Float16* __restrict__ W,
    const float* __restrict__ bias, const float* __restrict__ residual,
    void* __restrict__ out, int M, int L) {
  constexpr int K = HID;
  __shared__ __attribute__((aligned(16))) _Float16 As[2][64 * 64];
  const int lane = threadIdx.x & 31;
  const int wave = threadIdx.x >> 5;
  const int hlf  = lane >> 4;
  const int ln   = lane & 15;
  const int rowBase0 = blockIdx.y * 64;
  const int colBase  = blockIdx.x * 64;

  v8f acc[4] = {};

  // preload chunk 0
  if (wave == 0) {
    tdm_load_2d((u32)(size_t)(void*)&As[0][0], A + (size_t)rowBase0 * K,
                64, 64, (u32)K, (u32)(M - rowBase0), K);
  }

  int cur = 0;
  for (int k0 = 0; k0 < K; k0 += 64) {
    __syncthreads();  // all waves done reading As[cur^1] (overwritten next)
    if (wave == 0) {
      if (k0 + 64 < K) {
        tdm_load_2d((u32)(size_t)(void*)&As[cur ^ 1][0],
                    A + (size_t)rowBase0 * K + k0 + 64,
                    64, 64, (u32)(K - k0 - 64), (u32)(M - rowBase0), K);
        // in-order TDM: cnt<=1 means the chunk for As[cur] has landed
        __builtin_amdgcn_s_wait_tensorcnt(1);
      } else {
        __builtin_amdgcn_s_wait_tensorcnt(0);
      }
    }
    __syncthreads();  // As[cur] visible to all waves
#pragma unroll
    for (int ks = 0; ks < 2; ++ks) {
      const _Float16* arow = &As[cur][(wave * 16 + ln) * 64 + ks * 32];
      v16h af = load_a_frag(arow, hlf);
#pragma unroll
      for (int nt = 0; nt < 4; ++nt) {
        const _Float16* wrow =
            W + (size_t)(colBase + nt * 16 + ln) * K + k0 + ks * 32;
        acc[nt] = wmma16(af, load_b_frag(wrow, hlf), acc[nt]);
      }
    }
    if (hlf == 0 && k0 + 64 < K)  // stream next weight chunk toward L2/L0
      __builtin_prefetch(W + (size_t)(colBase + ln) * K + k0 + 64, 0, 1);
    cur ^= 1;
  }

  const int mbase = rowBase0 + wave * 16 + hlf * 8;  // C layout: M = r + hlf*8
#pragma unroll
  for (int nt = 0; nt < 4; ++nt) {
    const int n = colBase + nt * 16 + ln;
    const float bv = bias[n];
#pragma unroll
    for (int r = 0; r < 8; ++r) {
      const int m = mbase + r;
      float v = acc[nt][r] + bv;
      if constexpr (MODE == 2) {
        ((float*)out)[(size_t)m * HID + n] = v + residual[(size_t)m * HID + n];
      } else {
        const int b = m / L, l = m % L, h = n >> 6, d = n & 63;
        if constexpr (MODE == 0) {  // [B,H,L,64]
          ((_Float16*)out)[(((size_t)(b * HEADS + h) * L + l) << 6) + d] =
              (_Float16)v;
        } else {                    // [B,H,64,Lk] (V transposed for P@V)
          ((_Float16*)out)[((size_t)((b * HEADS + h) * 64 + d)) * LK + l] =
              (_Float16)v;
        }
      }
    }
  }
}

// ---------------------------------------------------------------------------
// Kernel 3: fused attention per (b, h, 16 q-rows). 256 threads = 8 waves.
//  phase 1: S = Q K^T * 1/8 + mask  (WMMA, scores -> LDS f32 [16][1024])
//  phase 2: block softmax (float4 LDS traffic); probs -> global f32 with
//           NON-TEMPORAL b128 stores (write-once stream, keep L2 clean);
//           P -> LDS f16
//  phase 3: ctx = P @ V (waves 0..3, WMMA, A-frags via ds_load_b128)
// ---------------------------------------------------------------------------
__global__ __launch_bounds__(256) void k_attention(
    const _Float16* __restrict__ Qh, const _Float16* __restrict__ Kh,
    const _Float16* __restrict__ Vt, const float* __restrict__ mask,
    float* __restrict__ probs, _Float16* __restrict__ ctx) {
  extern __shared__ char smem[];
  float*    sc     = (float*)smem;                          // 16*1024 f32 64KB
  _Float16* pb     = (_Float16*)(smem + 65536);             // 16*1024 f16 32KB
  float*    red    = (float*)(smem + 65536 + 32768);        // 16*16
  float*    rowmax = red + 256;                             // 16
  float*    rowinv = rowmax + 16;                           // 16

  const int qt = blockIdx.x, h = blockIdx.y, b = blockIdx.z;
  const int lane = threadIdx.x & 31, wave = threadIdx.x >> 5;
  const int hlf = lane >> 4, ln = lane & 15;
  const size_t bh = (size_t)(b * HEADS + h);

  // ---- phase 1: scores ----
  const _Float16* qrow = Qh + (bh * LQ + qt * 16 + ln) * HDIM;
  v16h aq0 = load_a_frag(qrow, hlf);        // d = 0..31
  v16h aq1 = load_a_frag(qrow + 32, hlf);   // d = 32..63
#pragma unroll 2
  for (int i = 0; i < 8; ++i) {
    const int kt = wave * 8 + i;            // 64 k-tiles over 8 waves
    const _Float16* krow = Kh + (bh * LK + kt * 16 + ln) * HDIM;
    v8f s = {};
    s = wmma16(aq0, load_b_frag(krow, hlf), s);
    s = wmma16(aq1, load_b_frag(krow + 32, hlf), s);
    const float mv = mask[(size_t)b * LK + kt * 16 + ln];
#pragma unroll
    for (int r = 0; r < 8; ++r)
      sc[(r + hlf * 8) * LK + kt * 16 + ln] = s[r] * 0.125f + mv;
  }
  __syncthreads();

  // ---- phase 2: softmax (16 rows x 1024; 16 threads/row, 16 float4 each) --
  {
    const int row = threadIdx.x >> 4;
    const int sub = threadIdx.x & 15;
    f32x4* srow4 = (f32x4*)(sc + (size_t)row * LK) + sub * 16;
    float pmax = -3.4e38f;
#pragma unroll 4
    for (int j = 0; j < 16; ++j) {
      f32x4 v = srow4[j];
      pmax = fmaxf(pmax, fmaxf(fmaxf(v.x, v.y), fmaxf(v.z, v.w)));
    }
    red[row * 16 + sub] = pmax;
    __syncthreads();
    if (sub == 0) {
      float m_ = red[row * 16];
      for (int j = 1; j < 16; ++j) m_ = fmaxf(m_, red[row * 16 + j]);
      rowmax[row] = m_;
    }
    __syncthreads();
    const float rm = rowmax[row];
    float psum = 0.f;
#pragma unroll 4
    for (int j = 0; j < 16; ++j) {
      f32x4 v = srow4[j];
      v.x = __expf(v.x - rm); v.y = __expf(v.y - rm);
      v.z = __expf(v.z - rm); v.w = __expf(v.w - rm);
      srow4[j] = v;
      psum += (v.x + v.y) + (v.z + v.w);
    }
    red[row * 16 + sub] = psum;
    __syncthreads();
    if (sub == 0) {
      float s_ = 0.f;
      for (int j = 0; j < 16; ++j) s_ += red[row * 16 + j];
      rowinv[row] = 1.0f / s_;
    }
    __syncthreads();
    const float ri = rowinv[row];
    f32x4* prow4 =
        (f32x4*)(probs + (bh * LQ + qt * 16 + row) * (size_t)LK) + sub * 16;
    v4h* pbrow4 = (v4h*)(pb + (size_t)row * LK) + sub * 16;
#pragma unroll 4
    for (int j = 0; j < 16; ++j) {
      f32x4 v = srow4[j];
      f32x4 p = {v.x * ri, v.y * ri, v.z * ri, v.w * ri};
      __builtin_nontemporal_store(p, prow4 + j);  // write-once stream -> NT
      v4h ph = {(_Float16)p.x, (_Float16)p.y, (_Float16)p.z, (_Float16)p.w};
      pbrow4[j] = ph;
    }
  }
  __syncthreads();

  // ---- phase 3: ctx = P(16x1024) @ V(1024x64), one 16-d tile per wave ----
  if (wave < 4) {
    v8f cacc = {};
    for (int kk0 = 0; kk0 < LK; kk0 += 32) {
      v16h ap = load_a_frag(pb + (size_t)ln * LK + kk0, hlf);  // ds_load_b128
      const _Float16* vrow = Vt + (bh * 64 + wave * 16 + ln) * (size_t)LK + kk0;
      cacc = wmma16(ap, load_b_frag(vrow, hlf), cacc);
    }
#pragma unroll
    for (int r = 0; r < 8; ++r) {
      const int q = qt * 16 + r + hlf * 8;
      ctx[((size_t)(b * LQ + q)) * HID + h * HDIM + wave * 16 + ln] =
          (_Float16)cacc[r];
    }
  }
}

// ---------------------------------------------------------------------------
// Kernel 4: LayerNorm over 768 per row (one block per row)
// ---------------------------------------------------------------------------
__global__ __launch_bounds__(256) void k_layernorm(
    const float* __restrict__ Hb, const float* __restrict__ gamma,
    const float* __restrict__ beta, float* __restrict__ out) {
  __shared__ float rs[256], rq[256];
  const int row = blockIdx.x, tid = threadIdx.x;
  const float* hrow = Hb + (size_t)row * HID;
  float s = 0.f, q = 0.f;
  for (int i = tid; i < HID; i += 256) {
    float v = hrow[i];
    s += v;
    q += v * v;
  }
  rs[tid] = s; rq[tid] = q;
  __syncthreads();
  for (int o = 128; o > 0; o >>= 1) {
    if (tid < o) { rs[tid] += rs[tid + o]; rq[tid] += rq[tid + o]; }
    __syncthreads();
  }
  const float mu   = rs[0] * (1.0f / HID);
  const float var  = rq[0] * (1.0f / HID) - mu * mu;
  const float rstd = rsqrtf(var + 1e-12f);
  float* orow = out + (size_t)row * HID;
  for (int i = tid; i < HID; i += 256)
    orow[i] = (hrow[i] - mu) * rstd * gamma[i] + beta[i];
}

// ---------------------------------------------------------------------------
extern "C" void kernel_launch(void* const* d_in, const int* in_sizes, int n_in,
                              void* d_out, int out_size, void* d_ws,
                              size_t ws_size, hipStream_t stream) {
  (void)in_sizes; (void)n_in; (void)out_size; (void)ws_size;
  const float* query = (const float*)d_in[0];
  const float* key   = (const float*)d_in[1];
  const float* value = (const float*)d_in[2];
  const float* amask = (const float*)d_in[3];
  const float* Wq = (const float*)d_in[5];
  const float* bq = (const float*)d_in[6];
  const float* Wk = (const float*)d_in[7];
  const float* bk = (const float*)d_in[8];
  const float* Wv = (const float*)d_in[9];
  const float* bv = (const float*)d_in[10];
  const float* Wo = (const float*)d_in[11];
  const float* bo = (const float*)d_in[12];
  const float* ln_g = (const float*)d_in[13];
  const float* ln_b = (const float*)d_in[14];

  float* out_ln = (float*)d_out;                               // [8,512,768]
  float* probs  = (float*)d_out + (size_t)BATCH * LQ * HID;    // [8,12,512,1024]

  // workspace carve-out (256B aligned regions), ~86.5 MB total
  char* ws = (char*)d_ws;
  size_t off = 0;
  auto carve = [&](size_t bytes) -> char* {
    char* p = ws + off;
    off += (bytes + 255) & ~(size_t)255;
    return p;
  };
  const size_t MQ = (size_t)BATCH * LQ;   // 4096
  const size_t MK = (size_t)BATCH * LK;   // 8192
  _Float16* Xq  = (_Float16*)carve(MQ * HID * 2);
  _Float16* Xk  = (_Float16*)carve(MK * HID * 2);
  _Float16* Xv  = (_Float16*)carve(MK * HID * 2);
  _Float16* Wqh = (_Float16*)carve((size_t)HID * HID * 2);
  _Float16* Wkh = (_Float16*)carve((size_t)HID * HID * 2);
  _Float16* Wvh = (_Float16*)carve((size_t)HID * HID * 2);
  _Float16* Woh = (_Float16*)carve((size_t)HID * HID * 2);
  _Float16* Qh  = (_Float16*)carve(MQ * HID * 2);   // [B,H,512,64]
  _Float16* Kh  = (_Float16*)carve(MK * HID * 2);   // [B,H,1024,64]
  _Float16* Vt  = (_Float16*)carve(MK * HID * 2);   // [B,H,64,1024]
  _Float16* Ctx = (_Float16*)carve(MQ * HID * 2);   // [B,512,768]
  float*    Hb  = (float*)carve(MQ * HID * 4);      // pre-LN f32

  // 1) convert f32 -> f16 (4-wide)
  const int NW4 = HID * HID / 4;
  k_f32_to_f16<<<2048, 256, 0, stream>>>(query, Xq, (int)(MQ * HID / 4));
  k_f32_to_f16<<<2048, 256, 0, stream>>>(key,   Xk, (int)(MK * HID / 4));
  k_f32_to_f16<<<2048, 256, 0, stream>>>(value, Xv, (int)(MK * HID / 4));
  k_f32_to_f16<<<1024, 256, 0, stream>>>(Wq, Wqh, NW4);
  k_f32_to_f16<<<1024, 256, 0, stream>>>(Wk, Wkh, NW4);
  k_f32_to_f16<<<1024, 256, 0, stream>>>(Wv, Wvh, NW4);
  k_f32_to_f16<<<1024, 256, 0, stream>>>(Wo, Woh, NW4);

  // 2) Q/K/V projections (WMMA + double-buffered TDM staging)
  k_gemm_wmma<0><<<dim3(HID / 64, (int)(MQ / 64)), 128, 0, stream>>>(
      Xq, Wqh, bq, nullptr, Qh, (int)MQ, LQ);
  k_gemm_wmma<0><<<dim3(HID / 64, (int)(MK / 64)), 128, 0, stream>>>(
      Xk, Wkh, bk, nullptr, Kh, (int)MK, LK);
  k_gemm_wmma<1><<<dim3(HID / 64, (int)(MK / 64)), 128, 0, stream>>>(
      Xv, Wvh, bv, nullptr, Vt, (int)MK, LK);

  // 3) fused attention (scores -> softmax -> NT probs out -> P@V)
  const size_t attn_smem = 65536 + 32768 + 256 * 4 + 16 * 4 + 16 * 4;
  k_attention<<<dim3(LQ / 16, HEADS, BATCH), 256, attn_smem, stream>>>(
      Qh, Kh, Vt, amask, probs, Ctx);

  // 4) output projection + residual, then LayerNorm -> d_out
  k_gemm_wmma<2><<<dim3(HID / 64, (int)(MQ / 64)), 128, 0, stream>>>(
      Ctx, Woh, bo, query, Hb, (int)MQ, LQ);
  k_layernorm<<<(int)MQ, 256, 0, stream>>>(Hb, ln_g, ln_b, out_ln);
}
